// WaveletDualAttentionBlock_46420006535711
// MI455X (gfx1250) — matrix operand order repaired and checked by reading, not compiled, not run on hardware
//
#include <hip/hip_runtime.h>
#include <math.h>

typedef __attribute__((ext_vector_type(16))) _Float16 v16h;
typedef __attribute__((ext_vector_type(8)))  _Float16 v8h;
typedef __attribute__((ext_vector_type(8)))  float    v8f;

#define B_     128
#define R_     14
#define N_     196
#define NT     208      // padded tokens (13*16)
#define DIM_   384
#define HEADS_ 8
#define HD_    48
#define HDP_   64       // padded per-head dim (2 x K=32 WMMA steps)
#define QKSTR  (HEADS_*HDP_)   // 512, q/k padded row stride (halves)
#define C3_    1152
#define KEYP_  224      // padded key dim for P x V (7*32)
#define SCALE_ 0.14433756729740643f   // 48^-0.5

// ---------------- WMMA fragment helpers (wave32, 16x16x32 f16) ----------------
// A (16x32, MxK): lane r=lane&15 holds row M=r; halves h<8 -> K=g*8+h at p[0..7],
// halves h>=8 -> K=16+g*8+(h-8) at p[16..23], where g=lane>>4 and p points at
// (row base + kb + g*8).
static __device__ __forceinline__ v16h ld_afrag(const _Float16* p) {
  v8h lo = *reinterpret_cast<const v8h*>(p);
  v8h hi = *reinterpret_cast<const v8h*>(p + 16);
  v16h r;
#pragma unroll
  for (int i = 0; i < 8; ++i) { r[i] = lo[i]; r[i + 8] = hi[i]; }
  return r;
}
// B (32x16, KxN): lane holds column N=lane&15; halves h<8 -> K=g*8+h, h>=8 ->
// K=16+g*8+(h-8), source stored N-major ([col][k]); p points at (col base + kb + g*8).
static __device__ __forceinline__ v16h ld_bfrag(const _Float16* p) {
  v8h lo = *reinterpret_cast<const v8h*>(p);
  v8h hi = *reinterpret_cast<const v8h*>(p + 16);
  v16h r;
#pragma unroll
  for (int i = 0; i < 8; ++i) { r[i] = lo[i]; r[i + 8] = hi[i]; }
  return r;
}
static __device__ __forceinline__ v8f zero8f() {
  v8f a;
#pragma unroll
  for (int i = 0; i < 8; ++i) a[i] = 0.0f;
  return a;
}
static __device__ __forceinline__ float gelu_f(float x) {
  return 0.5f * x * (1.0f + erff(x * 0.70710678118654752f));
}

// ---------------- small utility kernels ----------------
__global__ void k_f32to16(const float* __restrict__ s, _Float16* __restrict__ d, long n) {
  long i = (long)blockIdx.x * blockDim.x + threadIdx.x;
  if (i < n) d[i] = (_Float16)s[i];
}
__global__ void k_zfill16(_Float16* __restrict__ p, long n) {
  long i = (long)blockIdx.x * blockDim.x + threadIdx.x;
  if (i < n) p[i] = (_Float16)0.0f;
}

// RepVGGDW front: X0[b][tok][cc] = gelu(dw3(concat) + dw3_b + x*dw1_w + dw1_b + x), f16 token-major
__global__ void k_high_front(const float* __restrict__ lh, const float* __restrict__ hl,
                             const float* __restrict__ hh,
                             const float* __restrict__ dw3w, const float* __restrict__ dw3b,
                             const float* __restrict__ dw1w, const float* __restrict__ dw1b,
                             _Float16* __restrict__ X0) {
  long t = (long)blockIdx.x * blockDim.x + threadIdx.x;
  const long total = (long)B_ * NT * C3_;
  if (t >= total) return;
  int cc  = (int)(t % C3_);
  int tok = (int)((t / C3_) % NT);
  int b   = (int)(t / ((long)C3_ * NT));
  if (tok >= N_) { X0[t] = (_Float16)0.0f; return; }
  int t3 = cc / DIM_, ch = cc % DIM_;
  const float* src = (t3 == 0 ? lh : (t3 == 1 ? hl : hh)) + ((long)b * DIM_ + ch) * N_;
  int y = tok / R_, x = tok % R_;
  float center = src[tok];
  float conv = 0.0f;
#pragma unroll
  for (int ky = 0; ky < 3; ++ky) {
    int yy = y + ky - 1; if (yy < 0 || yy >= R_) continue;
#pragma unroll
    for (int kx = 0; kx < 3; ++kx) {
      int xx = x + kx - 1; if (xx < 0 || xx >= R_) continue;
      conv += src[yy * R_ + xx] * dw3w[cc * 9 + ky * 3 + kx];
    }
  }
  float val = conv + dw3b[cc] + center * dw1w[cc] + dw1b[cc] + center;
  X0[t] = (_Float16)gelu_f(val);
}

// EffectiveSE helpers
__global__ void k_mean(const float* __restrict__ ll, float* __restrict__ mean) {
  int i = blockIdx.x * blockDim.x + threadIdx.x;
  if (i >= B_ * DIM_) return;
  const float* p = ll + (long)i * N_;
  float s = 0.0f;
  for (int t = 0; t < N_; ++t) s += p[t];
  mean[i] = s * (1.0f / N_);
}
__global__ void k_gate(const float* __restrict__ mean, const float* __restrict__ w,
                       const float* __restrict__ bsrc, float* __restrict__ gate) {
  int i = blockIdx.x * blockDim.x + threadIdx.x;
  if (i >= B_ * DIM_) return;
  int b = i / DIM_, c = i % DIM_;
  const float* mv = mean + (long)b * DIM_;
  const float* wr = w + (long)c * DIM_;
  float s = bsrc[c];
  for (int k = 0; k < DIM_; ++k) s += wr[k] * mv[k];
  s = s * (1.0f / 6.0f) + 0.5f;
  gate[i] = fminf(fmaxf(s, 0.0f), 1.0f);
}
__global__ void k_se(const float* __restrict__ ll, const float* __restrict__ gate,
                     _Float16* __restrict__ se) {
  long t = (long)blockIdx.x * blockDim.x + threadIdx.x;
  const long total = (long)B_ * NT * DIM_;
  if (t >= total) return;
  int c   = (int)(t % DIM_);
  int tok = (int)((t / DIM_) % NT);
  int b   = (int)(t / ((long)DIM_ * NT));
  float v = 0.0f;
  if (tok < N_) v = ll[((long)b * DIM_ + c) * N_ + tok] * gate[b * DIM_ + c];
  se[t] = (_Float16)v;
}

// ---------------- generic WMMA GEMM over token-major activations ----------------
// Each wave computes a 16(token) x 64(cout) tile: one A fragment is reused by
// 4 B fragments / 4 accumulators per K step.  __launch_bounds__(32,1) gives the
// wave the full VGPR budget so the 5 fragment loads per step stay in distinct
// registers and pipeline ahead of the WMMAs.
enum { MODE_TOK16 = 0, MODE_F32CHW = 1, MODE_QKV_HIGH = 2, MODE_KV_LOW = 3, MODE_Q_LOW = 4 };

__global__ void __launch_bounds__(32, 1)
k_gemm_tok(const _Float16* __restrict__ W, const _Float16* __restrict__ X,
           const float* __restrict__ bias, const _Float16* __restrict__ resid,
           void* out0, void* out1, void* out2,
           int Cin, int Cout, int mode, int actGelu) {
  const int lane = threadIdx.x;
  const int r    = lane & 15;
  const int g    = lane >> 4;
  const int tokBase  = blockIdx.x * 16;
  const int coutBase = blockIdx.y * 64;
  const int b        = blockIdx.z;

  const _Float16* Xb = X + (long)b * NT * Cin;
  const _Float16* Ab = Xb + (long)(tokBase + r) * Cin;           // this lane's A row
  const _Float16* Wb[4];
#pragma unroll
  for (int j = 0; j < 4; ++j) Wb[j] = W + (long)(coutBase + j * 16 + r) * Cin;

  v8f acc0 = zero8f(), acc1 = zero8f(), acc2 = zero8f(), acc3 = zero8f();
  for (int kb = 0; kb < Cin; kb += 32) {
    // WGP-scope prefetch of the next weight K-slab (global_prefetch_b8)
    __builtin_prefetch(Wb[0] + kb + 32, 0, 3);
    v16h a  = ld_afrag(Ab + kb + g * 8);
    v16h b0 = ld_bfrag(Wb[0] + kb + g * 8);
    v16h b1 = ld_bfrag(Wb[1] + kb + g * 8);
    v16h b2 = ld_bfrag(Wb[2] + kb + g * 8);
    v16h b3 = ld_bfrag(Wb[3] + kb + g * 8);
    acc0 = __builtin_amdgcn_wmma_f32_16x16x32_f16(false, a, false, b0, (short)0, acc0, false, false);
    acc1 = __builtin_amdgcn_wmma_f32_16x16x32_f16(false, a, false, b1, (short)0, acc1, false, false);
    acc2 = __builtin_amdgcn_wmma_f32_16x16x32_f16(false, a, false, b2, (short)0, acc2, false, false);
    acc3 = __builtin_amdgcn_wmma_f32_16x16x32_f16(false, a, false, b3, (short)0, acc3, false, false);
  }

#pragma unroll
  for (int j = 0; j < 4; ++j) {
    v8f acc = (j == 0) ? acc0 : (j == 1) ? acc1 : (j == 2) ? acc2 : acc3;
    const int cc = coutBase + j * 16 + r;   // this lane's output column
#pragma unroll
    for (int v = 0; v < 8; ++v) {
      int tok = tokBase + g * 8 + v;
      float val = acc[v] + bias[cc];
      if (resid) val += (float)resid[((long)b * NT + tok) * Cout + cc];
      if (actGelu) val = gelu_f(val);
      if (mode == MODE_TOK16) {
        ((_Float16*)out0)[((long)b * NT + tok) * Cout + cc] = (_Float16)val;
      } else if (mode == MODE_F32CHW) {
        if (tok < N_) ((float*)out0)[((long)b * Cout + cc) * N_ + tok] = val;
      } else if (mode == MODE_QKV_HIGH) {
        int t3 = cc / DIM_, rr = cc % DIM_;
        int hh = rr / HD_, dd = rr % HD_;
        if (t3 == 0)
          ((_Float16*)out0)[((long)b * NT + tok) * QKSTR + hh * HDP_ + dd] = (_Float16)val;
        else if (t3 == 1)
          ((_Float16*)out1)[((long)b * NT + tok) * QKSTR + hh * HDP_ + dd] = (_Float16)val;
        else
          ((_Float16*)out2)[(long)b * HEADS_ * HDP_ * KEYP_ + (hh * HDP_ + dd) * KEYP_ + tok] =
              (_Float16)val;
      } else if (mode == MODE_KV_LOW) {
        int rr = (cc < DIM_) ? cc : cc - DIM_;
        int hh = rr / HD_, dd = rr % HD_;
        if (cc < DIM_)
          ((_Float16*)out0)[((long)b * NT + tok) * QKSTR + hh * HDP_ + dd] = (_Float16)val;
        else
          ((_Float16*)out2)[(long)b * HEADS_ * HDP_ * KEYP_ + (hh * HDP_ + dd) * KEYP_ + tok] =
              (_Float16)val;
      } else { // MODE_Q_LOW
        int hh = cc / HD_, dd = cc % HD_;
        ((_Float16*)out0)[((long)b * NT + tok) * QKSTR + hh * HDP_ + dd] = (_Float16)val;
      }
    }
  }
}

// ---------------- attention: one wave per (qtile, head, batch) ----------------
// The V head slab (48x224 f16, contiguous in vt) is streamed into LDS with
// global_load_async_to_lds_b128 at wave start; the copies overlap with the
// QK^T WMMAs and the softmax, then s_wait_asynccnt 0 gates phase 3, whose B
// fragments come from LDS (ds_load_b128).
__global__ void __launch_bounds__(32, 1)
k_attn(const _Float16* __restrict__ qpad, const _Float16* __restrict__ kpad,
       const _Float16* __restrict__ vt, const float* __restrict__ biasTab,
       const int* __restrict__ bidx, _Float16* __restrict__ out) {
  __shared__ __align__(16) float    Sb[16][KEYP_];
  __shared__ __align__(16) _Float16 Pb[16][KEYP_];
  __shared__ __align__(16) _Float16 Vs[HD_ * KEYP_];   // 48x224 f16 = 21504 B
  const int lane = threadIdx.x, r = lane & 15, g = lane >> 4;
  const int qBase = blockIdx.x * 16, h = blockIdx.y, b = blockIdx.z;
  const _Float16* qb = qpad + (long)b * NT * QKSTR + h * HDP_;
  const _Float16* kb = kpad + (long)b * NT * QKSTR + h * HDP_;
  const _Float16* vb = vt + (long)b * HEADS_ * HDP_ * KEYP_ + (long)h * HDP_ * KEYP_;

  // Kick off async V copy: 48*224*2 = 21504 B = 42 iters * 32 lanes * 16 B
  {
    const unsigned lbase = (unsigned)(uintptr_t)&Vs[0];   // low 32 bits = LDS offset
    const char* gsrc = (const char*)vb;
#pragma unroll 1
    for (int it = 0; it < 42; ++it) {
      int idx = it * 32 + lane;
      unsigned laddr = lbase + idx * 16;
      const char* ga = gsrc + idx * 16;
      asm volatile("global_load_async_to_lds_b128 %0, %1, off"
                   :: "v"(laddr), "v"(ga) : "memory");
    }
  }

  // Phase 1: S = (Q K^T)*scale + bias.  Q fragments are invariant across the
  // 13 key tiles -> load once, reuse (A-frag hoisting).
  v16h aq0 = ld_afrag(qb + (long)(qBase + r) * QKSTR + 0  + g * 8);
  v16h aq1 = ld_afrag(qb + (long)(qBase + r) * QKSTR + 32 + g * 8);
  for (int kt = 0; kt < 13; ++kt) {
    v8f acc = zero8f();
    v16h b0 = ld_bfrag(kb + (long)(kt * 16 + r) * QKSTR + 0  + g * 8);
    v16h b1 = ld_bfrag(kb + (long)(kt * 16 + r) * QKSTR + 32 + g * 8);
    acc = __builtin_amdgcn_wmma_f32_16x16x32_f16(false, aq0, false, b0, (short)0, acc, false, false);
    acc = __builtin_amdgcn_wmma_f32_16x16x32_f16(false, aq1, false, b1, (short)0, acc, false, false);
#pragma unroll
    for (int v = 0; v < 8; ++v) {
      int row = qBase + g * 8 + v;
      int col = kt * 16 + r;
      float s;
      if (col < N_) {
        float bv = 0.0f;
        if (row < N_) bv = biasTab[h * N_ + bidx[row * N_ + col]];
        s = acc[v] * SCALE_ + bv;
      } else {
        s = -1e30f;   // mask padded key columns
      }
      Sb[g * 8 + v][col] = s;
    }
  }
  __syncthreads();

  // Phase 2: row softmax (lane L owns row L), write P (f16) padded to 224 keys
  if (lane < 16) {
    float m = -1e30f;
    for (int c = 0; c < NT; ++c) m = fmaxf(m, Sb[lane][c]);
    float sum = 0.0f;
    for (int c = 0; c < NT; ++c) { float e = __expf(Sb[lane][c] - m); Sb[lane][c] = e; sum += e; }
    float inv = 1.0f / sum;
    for (int c = 0; c < NT; ++c)     Pb[lane][c] = (_Float16)(Sb[lane][c] * inv);
    for (int c = NT; c < KEYP_; ++c) Pb[lane][c] = (_Float16)0.0f;
  }
  asm volatile("s_wait_asynccnt 0" ::: "memory");   // V slab resident in LDS
  __syncthreads();

  // Phase 3: O = P x V^T entirely out of LDS. One P fragment is reused by the
  // 3 d-column tiles per K-chunk (3 accumulators).
  v8f o0 = zero8f(), o1 = zero8f(), o2 = zero8f();
  for (int kk = 0; kk < KEYP_; kk += 32) {
    v16h a  = ld_afrag(&Pb[r][kk + g * 8]);
    v16h b0 = ld_bfrag(&Vs[(0 * 16 + r) * KEYP_ + kk + g * 8]);
    v16h b1 = ld_bfrag(&Vs[(1 * 16 + r) * KEYP_ + kk + g * 8]);
    v16h b2 = ld_bfrag(&Vs[(2 * 16 + r) * KEYP_ + kk + g * 8]);
    o0 = __builtin_amdgcn_wmma_f32_16x16x32_f16(false, a, false, b0, (short)0, o0, false, false);
    o1 = __builtin_amdgcn_wmma_f32_16x16x32_f16(false, a, false, b1, (short)0, o1, false, false);
    o2 = __builtin_amdgcn_wmma_f32_16x16x32_f16(false, a, false, b2, (short)0, o2, false, false);
  }
#pragma unroll
  for (int ct = 0; ct < 3; ++ct) {
    v8f acc = (ct == 0) ? o0 : (ct == 1) ? o1 : o2;
#pragma unroll
    for (int v = 0; v < 8; ++v) {
      int tok = qBase + g * 8 + v;
      int c   = h * HD_ + ct * 16 + r;
      out[((long)b * NT + tok) * DIM_ + c] = (_Float16)acc[v];
    }
  }
}

// ---------------- host orchestration ----------------
extern "C" void kernel_launch(void* const* d_in, const int* in_sizes, int n_in,
                              void* d_out, int out_size, void* d_ws, size_t ws_size,
                              hipStream_t stream) {
  (void)in_sizes; (void)n_in; (void)out_size;
  char* ws = (char*)d_ws;
  size_t o = 0;
  auto take = [&](size_t bytes) { size_t r = o; o += (bytes + 255) & ~(size_t)255; return r; };

  // f16 weight copies
  const int    widx[10] = {8, 10, 12, 15, 17, 21, 23, 25, 28, 30};
  const size_t wel[10]  = {(size_t)DIM_ * C3_, (size_t)C3_ * DIM_, (size_t)DIM_ * DIM_,
                           (size_t)768 * DIM_, (size_t)DIM_ * 768, (size_t)DIM_ * DIM_,
                           (size_t)768 * DIM_, (size_t)DIM_ * DIM_, (size_t)768 * DIM_,
                           (size_t)DIM_ * 768};
  size_t off_w[10];
  for (int i = 0; i < 10; ++i) off_w[i] = take(wel[i] * 2);

  const size_t X0B   = (size_t)B_ * NT * C3_ * 2;
  const size_t ACTB  = (size_t)B_ * NT * DIM_ * 2;        // [B][208][384] f16
  const size_t QKB   = (size_t)B_ * NT * QKSTR * 2;       // [B][208][512] f16
  const size_t VTB   = (size_t)B_ * HEADS_ * HDP_ * KEYP_ * 2;
  const size_t T2B   = (size_t)B_ * NT * 768 * 2;

  size_t off_X0 = take(X0B);            // X0 reused for qpad+kpad after GEMM1
  size_t off_qpad = off_X0;
  size_t off_kpad = off_X0 + QKB;       // qpad+kpad (54.5MB) fits inside X0 (61.3MB)
  size_t off_hc  = take(ACTB);          // hc, later reused for se
  size_t off_vt  = take(VTB);
  size_t off_ao  = take(ACTB);          // attention output (both branches)
  size_t off_H1  = take(ACTB);          // high_attn_out, later reused for ll_attn
  size_t off_T   = take(T2B);           // FFN hidden (both branches)
  size_t off_RH  = take(ACTB);          // refined_high
  size_t off_mean = take((size_t)B_ * DIM_ * 4);
  size_t off_gate = take((size_t)B_ * DIM_ * 4);
  if (ws_size < o) return;              // workspace too small; nothing safe to do

  _Float16* w16[10];
  for (int i = 0; i < 10; ++i) w16[i] = (_Float16*)(ws + off_w[i]);
  _Float16* X0   = (_Float16*)(ws + off_X0);
  _Float16* qpad = (_Float16*)(ws + off_qpad);
  _Float16* kpad = (_Float16*)(ws + off_kpad);
  _Float16* hc   = (_Float16*)(ws + off_hc);
  _Float16* se   = hc;
  _Float16* vt   = (_Float16*)(ws + off_vt);
  _Float16* ao   = (_Float16*)(ws + off_ao);
  _Float16* H1   = (_Float16*)(ws + off_H1);
  _Float16* L1   = H1;
  _Float16* T    = (_Float16*)(ws + off_T);
  _Float16* RH   = (_Float16*)(ws + off_RH);
  float* meanb = (float*)(ws + off_mean);
  float* gateb = (float*)(ws + off_gate);

  const float* ll = (const float*)d_in[0];
  const float* lh = (const float*)d_in[1];
  const float* hl = (const float*)d_in[2];
  const float* hh = (const float*)d_in[3];
  const int*   bidx = (const int*)d_in[32];

  auto gl = [](long n) { return dim3((unsigned)((n + 255) / 256)); };

  // 0) weights -> f16
  for (int i = 0; i < 10; ++i)
    k_f32to16<<<gl((long)wel[i]), 256, 0, stream>>>((const float*)d_in[widx[i]], w16[i],
                                                    (long)wel[i]);
  // 1) RepVGGDW + GELU -> X0 (token-major f16)
  k_high_front<<<gl((long)B_ * NT * C3_), 256, 0, stream>>>(
      lh, hl, hh, (const float*)d_in[4], (const float*)d_in[5], (const float*)d_in[6],
      (const float*)d_in[7], X0);
  // 2) hc = pw_w x X0 + pw_b
  k_gemm_tok<<<dim3(13, DIM_ / 64, B_), 32, 0, stream>>>(
      w16[0], X0, (const float*)d_in[9], nullptr, hc, nullptr, nullptr, C3_, DIM_,
      MODE_TOK16, 0);
  // 3) zero q/k pads (reuses X0 region) + vt pads
  k_zfill16<<<gl(2 * (long)B_ * NT * QKSTR), 256, 0, stream>>>(qpad, 2 * (long)B_ * NT * QKSTR);
  k_zfill16<<<gl((long)B_ * HEADS_ * HDP_ * KEYP_), 256, 0, stream>>>(
      vt, (long)B_ * HEADS_ * HDP_ * KEYP_);
  // 4) qkv = hqkv_w x hc  (scatter q->qpad, k->kpad, v->vt)
  k_gemm_tok<<<dim3(13, C3_ / 64, B_), 32, 0, stream>>>(
      w16[1], hc, (const float*)d_in[11], nullptr, qpad, kpad, vt, DIM_, C3_,
      MODE_QKV_HIGH, 0);
  // 5) high MHSA
  k_attn<<<dim3(13, HEADS_, B_), 32, 0, stream>>>(qpad, kpad, vt, (const float*)d_in[14],
                                                  bidx, ao);
  // 6) H1 = hc + hproj x ao
  k_gemm_tok<<<dim3(13, DIM_ / 64, B_), 32, 0, stream>>>(
      w16[2], ao, (const float*)d_in[13], hc, H1, nullptr, nullptr, DIM_, DIM_,
      MODE_TOK16, 0);
  // 7) T = gelu(hm1 x H1)
  k_gemm_tok<<<dim3(13, 768 / 64, B_), 32, 0, stream>>>(
      w16[3], H1, (const float*)d_in[16], nullptr, T, nullptr, nullptr, DIM_, 768,
      MODE_TOK16, 1);
  // 8) RH = H1 + hm2 x T
  k_gemm_tok<<<dim3(13, DIM_ / 64, B_), 32, 0, stream>>>(
      w16[4], T, (const float*)d_in[18], H1, RH, nullptr, nullptr, 768, DIM_,
      MODE_TOK16, 0);
  // 9) EffectiveSE -> se (f16 token-major; reuses hc)
  k_mean<<<gl((long)B_ * DIM_), 256, 0, stream>>>(ll, meanb);
  k_gate<<<gl((long)B_ * DIM_), 256, 0, stream>>>(meanb, (const float*)d_in[19],
                                                  (const float*)d_in[20], gateb);
  k_se<<<gl((long)B_ * NT * DIM_), 256, 0, stream>>>(ll, gateb, se);
  // 10) lq = lq_w x se -> qpad ; kv = lkv_w x RH -> kpad/vt (d-pads still zero)
  k_gemm_tok<<<dim3(13, DIM_ / 64, B_), 32, 0, stream>>>(
      w16[5], se, (const float*)d_in[22], nullptr, qpad, nullptr, nullptr, DIM_, DIM_,
      MODE_Q_LOW, 0);
  k_gemm_tok<<<dim3(13, 768 / 64, B_), 32, 0, stream>>>(
      w16[6], RH, (const float*)d_in[24], nullptr, kpad, nullptr, vt, DIM_, 768,
      MODE_KV_LOW, 0);
  // 11) low cross-attention
  k_attn<<<dim3(13, HEADS_, B_), 32, 0, stream>>>(qpad, kpad, vt, (const float*)d_in[27],
                                                  bidx, ao);
  // 12) L1 = se + lproj x ao
  k_gemm_tok<<<dim3(13, DIM_ / 64, B_), 32, 0, stream>>>(
      w16[7], ao, (const float*)d_in[26], se, L1, nullptr, nullptr, DIM_, DIM_,
      MODE_TOK16, 0);
  // 13) T = gelu(lm1 x L1)
  k_gemm_tok<<<dim3(13, 768 / 64, B_), 32, 0, stream>>>(
      w16[8], L1, (const float*)d_in[29], nullptr, T, nullptr, nullptr, DIM_, 768,
      MODE_TOK16, 1);
  // 14) out = L1 + lm2 x T  -> fp32 channel-major [B,384,14,14]
  k_gemm_tok<<<dim3(13, DIM_ / 64, B_), 32, 0, stream>>>(
      w16[9], T, (const float*)d_in[31], L1, d_out, nullptr, nullptr, 768, DIM_,
      MODE_F32CHW, 0);
}
